// SingleHeadAttention_61529701482667
// MI455X (gfx1250) — matrix operand order repaired and checked by reading, not compiled, not run on hardware
//
#include <hip/hip_runtime.h>

// ---------------------------------------------------------------------------
// Types / helpers
// ---------------------------------------------------------------------------
typedef __bf16 bf16_t;
typedef __attribute__((ext_vector_type(16))) __bf16 v16bf;
typedef __attribute__((ext_vector_type(8)))  __bf16 v8bf;
typedef __attribute__((ext_vector_type(8)))  float  v8f;

#define WMMA_BF16(a, b, c) \
    __builtin_amdgcn_wmma_f32_16x16x32_bf16(false, (a), false, (b), (short)0, (c), false, false)

#define CAT16(lo, hi) __builtin_shufflevector((lo), (hi), 0, 1, 2, 3, 4, 5, 6, 7, \
                                              8, 9, 10, 11, 12, 13, 14, 15)

static __device__ __forceinline__ unsigned lds_addr(const void* p) {
    return (unsigned)(size_t)((__attribute__((address_space(3))) const char*)p);
}
static __device__ __forceinline__ void async_cp16(unsigned lds, const void* g) {
    asm volatile("global_load_async_to_lds_b128 %0, %1, off"
                 :: "v"(lds), "v"((unsigned long long)g) : "memory");
}

// Problem constants
constexpr int Bb = 4;
constexpr int S  = 2048;
constexpr int D  = 1024;
constexpr int M  = Bb * S;        // 8192 rows for the projection GEMMs
constexpr float SCALE = 0.03125f; // 1/sqrt(1024)
constexpr int LDSTR = 40;         // padded LDS row stride (elems): 20 banks

// Workspace layout (bf16 elements)
constexpr size_t XH_OFF  = 0;                       // x as bf16       [M x D]
constexpr size_t WH_OFF  = (size_t)M * D;           // Wq,Wk,Wv bf16   3x[D x D]
constexpr size_t QKV_OFF = WH_OFF + (size_t)3 * D * D;
// planes at QKV_OFF: Q [M x D], K [M x D], V^T [D x M]

// ---------------------------------------------------------------------------
// Kernel 0: one-time f32 -> bf16 conversion (8 elems/thread)
// ---------------------------------------------------------------------------
__global__ __launch_bounds__(256)
void sha_cvt_bf16_kernel(const float* __restrict__ src,
                         bf16_t* __restrict__ dst, int n8) {
    const int i = blockIdx.x * 256 + threadIdx.x;
    if (i < n8) {
        v8f f = *(const v8f*)(src + (size_t)i * 8);
        *(v8bf*)(dst + (size_t)i * 8) = __builtin_convertvector(f, v8bf);
    }
}

// ---------------------------------------------------------------------------
// Kernel 1: fused QKV projection, LDS-tiled bf16 GEMM.
//   out_z[m][e] = sum_d xh[m][d] * Wh_z[e][d]
// grid = (64*8, 1, 3), block = 256 (8 waves).  Block -> 128x128 output tile.
// Per K-step: 8KB A-slab + 8KB B-slab async-DMA'd to LDS (double-buffered,
// ASYNCcnt throttled), each wave computes a 64x32 sub-tile (8 WMMAs) from
// LDS fragments.  z==2 (V) is stored TRANSPOSED [D x M].
// ---------------------------------------------------------------------------
__global__ __launch_bounds__(256)
void sha_qkv_proj_kernel(const bf16_t* __restrict__ xh,
                         const bf16_t* __restrict__ wh,   // 3 planes [D x D]
                         bf16_t* __restrict__ qp,
                         bf16_t* __restrict__ kp,
                         bf16_t* __restrict__ vtp) {      // V^T [D x M]
    __shared__ __align__(16) bf16_t Ash[2][128 * LDSTR];  // 2 x 10 KB
    __shared__ __align__(16) bf16_t Bsh[2][128 * LDSTR];  // 2 x 10 KB

    const int z = blockIdx.z;
    const bf16_t* __restrict__ W = wh + (size_t)z * D * D;

    const int m0  = (blockIdx.x >> 3) * 128;
    const int nb0 = (blockIdx.x & 7) * 128;

    const int tid  = threadIdx.x;
    const int lane = tid & 31;
    const int wave = tid >> 5;
    const int wr   = wave >> 2;                 // 0..1 -> 64-row half
    const int wc   = wave & 3;                  // 0..3 -> 32-col strip

    const int ln16  = lane & 15;
    const int khalf = (lane >> 4) * 8;
    const int kb16  = (lane >> 4) * 16;

    const unsigned aB[2] = { lds_addr(Ash[0]), lds_addr(Ash[1]) };
    const unsigned bB[2] = { lds_addr(Bsh[0]), lds_addr(Bsh[1]) };

    // Per-thread staging role: 4 x 16B chunks per slab pair
    const int      srow = tid >> 2;             // 0..63
    const int      part = tid & 3;              // 16B chunk within 64B row
    const unsigned loff = (unsigned)part * 16u;
    const bf16_t* __restrict__ gA = xh + (size_t)m0 * D + part * 8;
    const bf16_t* __restrict__ gB = W + (size_t)nb0 * D + part * 8;

    auto stage = [&](int buf, int k0) {
        const bf16_t* a = gA + k0;
        const bf16_t* b = gB + k0;
        async_cp16(aB[buf] + (unsigned)srow * (LDSTR * 2) + loff,        a + (size_t)srow * D);
        async_cp16(aB[buf] + (unsigned)(64 + srow) * (LDSTR * 2) + loff, a + (size_t)(64 + srow) * D);
        async_cp16(bB[buf] + (unsigned)srow * (LDSTR * 2) + loff,        b + (size_t)srow * D);
        async_cp16(bB[buf] + (unsigned)(64 + srow) * (LDSTR * 2) + loff, b + (size_t)(64 + srow) * D);
    };

    v8f acc[4][2] = {};

    stage(0, 0);
    for (int ks = 0; ks < 32; ++ks) {
        const int cur = ks & 1;
        if (ks + 1 < 32) {
            stage(cur ^ 1, (ks + 1) * 32);
            asm volatile("s_wait_asynccnt 0x4" ::: "memory");   // cur slab done
        } else {
            asm volatile("s_wait_asynccnt 0x0" ::: "memory");
        }
        __syncthreads();

        const bf16_t* __restrict__ Ab = Ash[cur];
        const bf16_t* __restrict__ Bl = Bsh[cur];

        v16bf a[4];
#pragma unroll
        for (int fi = 0; fi < 4; ++fi) {
            const int rl = wr * 64 + fi * 16 + ln16;
            v8bf lo = *(const v8bf*)(Ab + rl * LDSTR + khalf);
            v8bf hi = *(const v8bf*)(Ab + rl * LDSTR + 16 + khalf);
            a[fi] = CAT16(lo, hi);
        }
        v16bf bb[2];
#pragma unroll
        for (int fj = 0; fj < 2; ++fj) {
            const int cl = wc * 32 + fj * 16 + ln16;
            v8bf lo = *(const v8bf*)(Bl + cl * LDSTR + khalf);
            v8bf hi = *(const v8bf*)(Bl + cl * LDSTR + 16 + khalf);
            bb[fj] = CAT16(lo, hi);
        }
#pragma unroll
        for (int fi = 0; fi < 4; ++fi)
#pragma unroll
            for (int fj = 0; fj < 2; ++fj)
                acc[fi][fj] = WMMA_BF16(a[fi], bb[fj], acc[fi][fj]);

        __syncthreads();
    }

    // C layout: elem r -> row (lane>=16)*8 + r, col = ln16
    if (z != 2) {
        bf16_t* __restrict__ out = (z == 0) ? qp : kp;
#pragma unroll
        for (int fi = 0; fi < 4; ++fi) {
            const int mrow = m0 + wr * 64 + fi * 16 + (lane >> 4) * 8;
#pragma unroll
            for (int fj = 0; fj < 2; ++fj) {
                const int col = nb0 + wc * 32 + fj * 16 + ln16;
                v8bf ob = __builtin_convertvector(acc[fi][fj], v8bf);
#pragma unroll
                for (int r = 0; r < 8; ++r)
                    out[(size_t)(mrow + r) * D + col] = ob[r];
            }
        }
    } else {
        // Transposed store: Vt[e][token]; accumulator rows are a contiguous
        // token run -> single 16B vector store per tile.
#pragma unroll
        for (int fi = 0; fi < 4; ++fi) {
            const int mrow = m0 + wr * 64 + fi * 16 + (lane >> 4) * 8;
#pragma unroll
            for (int fj = 0; fj < 2; ++fj) {
                const int col = nb0 + wc * 32 + fj * 16 + ln16;
                v8bf ob = __builtin_convertvector(acc[fi][fj], v8bf);
                *(v8bf*)(vtp + (size_t)col * M + mrow) = ob;
            }
        }
    }
}

// ---------------------------------------------------------------------------
// Kernel 2: block-cooperative flash attention with causal + padding mask.
// grid = (S/16, B), block = 256 (8 waves), ~52 KB static LDS.
//   - 32 KB Q tile staged ONCE per block via global_load_async_to_lds_b128
//   - kdim of Q.K^T split across the 8 waves (128 dims each), LDS tree-reduce
//   - online softmax (running max / sum / alpha) in f32
//   - P.V reads V^T so B-fragments are contiguous 32B global loads
// ---------------------------------------------------------------------------
__global__ __launch_bounds__(256)
void sha_flash_attn_kernel(const bf16_t* __restrict__ Qb,
                           const bf16_t* __restrict__ Kb,
                           const bf16_t* __restrict__ Vt,   // [D x M]
                           const int*    __restrict__ attn_mask,
                           float*        __restrict__ out) {
    __shared__ float  s_part[8 * 512];                 // 16 KB
    __shared__ float  s_S[512];                        //  2 KB
    __shared__ __align__(16) bf16_t s_P[512];          //  1 KB
    __shared__ __align__(32) float  s_mrun[16];
    __shared__ __align__(32) float  s_lrun[16];
    __shared__ __align__(32) float  s_alpha[16];
    __shared__ __align__(16) bf16_t Qlds[16 * 1024];   // 32 KB

    const int b    = blockIdx.y;
    const int q0   = blockIdx.x * 16;
    const int rowb = b * S;

    const int tid  = threadIdx.x;
    const int lane = tid & 31;
    const int wave = tid >> 5;

    const int ln16  = lane & 15;
    const int khalf = (lane >> 4) * 8;
    const int kb16  = (lane >> 4) * 16;

    // ---- stage the 32 KB Q tile (contiguous in global) via async DMA ------
    {
        const unsigned qbase = lds_addr(Qlds);
        const char* qsrc = (const char*)(Qb + (size_t)(rowb + q0) * D);
#pragma unroll
        for (int c = tid; c < 2048; c += 256)          // 2048 x 16B chunks
            async_cp16(qbase + (unsigned)c * 16u, qsrc + (size_t)c * 16u);
    }

    if (tid < 16) { s_mrun[tid] = -1e30f; s_lrun[tid] = 0.0f; }

    v8f o[8] = {};                          // 16 x 128 f32 output slice
    const int d0     = wave * 128;
    const int kslice = wave * 128;

    asm volatile("s_wait_asynccnt 0x0" ::: "memory");
    __syncthreads();

    const int nkt = (q0 + 16 + 31) >> 5;    // 32-key tiles covering [0, q0+16)

    for (int kt = 0; kt < nkt; ++kt) {
        const int key0 = kt * 32;

        // ---- 1) partial scores over this wave's 128-dim slice -------------
        v8f c0 = {}, c1 = {};
#pragma unroll
        for (int ks = 0; ks < 4; ++ks) {
            const int k0 = kslice + ks * 32;
            v8bf lo = *(const v8bf*)(Qlds + ln16 * D + k0 + kb16);
            v8bf hi = *(const v8bf*)(Qlds + ln16 * D + k0 + kb16 + 8);
            v16bf a = CAT16(lo, hi);
            v16bf b0 = *(const v16bf*)(Kb + (size_t)(rowb + key0 + ln16) * D + k0 + kb16);
            v16bf b1 = *(const v16bf*)(Kb + (size_t)(rowb + key0 + 16 + ln16) * D + k0 + kb16);
            c0 = WMMA_BF16(a, b0, c0);
            c1 = WMMA_BF16(a, b1, c1);
        }
#pragma unroll
        for (int r = 0; r < 8; ++r) {
            const int m = (lane >> 4) * 8 + r;
            s_part[wave * 512 + m * 32 + ln16]      = c0[r];
            s_part[wave * 512 + m * 32 + 16 + ln16] = c1[r];
        }
        __syncthreads();

        // ---- 2) tree-reduce partials, apply masks + 1/sqrt(D) -------------
        for (int idx = tid; idx < 512; idx += 256) {
            float s = 0.0f;
#pragma unroll
            for (int ww = 0; ww < 8; ++ww) s += s_part[ww * 512 + idx];
            const int m   = idx >> 5;
            const int col = idx & 31;
            const int key = key0 + col;
            const bool masked = (key > q0 + m) || (attn_mask[b * S + key] == 0);
            s_S[idx] = masked ? -1e30f : s * SCALE;
        }
        __syncthreads();

        // ---- 3) online softmax update (16 threads per row) ----------------
        {
            const int row = tid >> 4;
            const int c   = tid & 15;
            float s1 = s_S[row * 32 + c];
            float s2 = s_S[row * 32 + 16 + c];
            float tmax = fmaxf(s1, s2);
#pragma unroll
            for (int off = 8; off >= 1; off >>= 1)
                tmax = fmaxf(tmax, __shfl_xor(tmax, off, 32));
            const float mold = s_mrun[row];
            const float mnew = fmaxf(mold, tmax);
            const float p1 = __expf(s1 - mnew);
            const float p2 = __expf(s2 - mnew);
            float ts = p1 + p2;
#pragma unroll
            for (int off = 8; off >= 1; off >>= 1)
                ts += __shfl_xor(ts, off, 32);
            if (c == 0) {
                const float alpha = __expf(mold - mnew);
                s_lrun[row] = s_lrun[row] * alpha + ts;
                s_mrun[row] = mnew;
                s_alpha[row] = alpha;
            }
            s_P[row * 32 + c]      = (bf16_t)p1;
            s_P[row * 32 + 16 + c] = (bf16_t)p2;
        }
        __syncthreads();

        // ---- 4) rescale accumulators, P.V on this wave's 128-wide slice ---
        v8bf plo = *(const v8bf*)(s_P + ln16 * 32 + kb16);
        v8bf phi = *(const v8bf*)(s_P + ln16 * 32 + kb16 + 8);
        v16bf pfrag = CAT16(plo, phi);
        v8f ar = *(const v8f*)&s_alpha[(lane >> 4) * 8];

#pragma unroll
        for (int t = 0; t < 8; ++t) {
            const int dcol = d0 + t * 16 + ln16;
            // V^T: keys are contiguous -> single 32B fragment load
            v16bf vf = *(const v16bf*)(Vt + (size_t)dcol * M + rowb + key0 + kb16);
#pragma unroll
            for (int r = 0; r < 8; ++r) o[t][r] *= ar[r];
            o[t] = WMMA_BF16(pfrag, vf, o[t]);
        }

        // prefetch next K / V^T tiles (global_prefetch_b8)
        if (kt + 1 < nkt) {
            __builtin_prefetch(Kb + (size_t)(rowb + key0 + 32 + lane) * D + kslice, 0, 1);
            __builtin_prefetch(Vt + (size_t)(d0 + (lane & 15)) * M + rowb + key0 + 32, 0, 1);
        }
        __syncthreads();
    }

    // ---- epilogue: normalize (v_rcp_f32 instead of 64 divides), store f32 -
    v8f lv = *(const v8f*)&s_lrun[(lane >> 4) * 8];
    float inv[8];
#pragma unroll
    for (int r = 0; r < 8; ++r) inv[r] = __builtin_amdgcn_rcpf(lv[r]);
#pragma unroll
    for (int t = 0; t < 8; ++t) {
        const int dcol = d0 + t * 16 + ln16;
#pragma unroll
        for (int r = 0; r < 8; ++r) {
            const int m = (lane >> 4) * 8 + r;
            out[(size_t)(rowb + q0 + m) * D + dcol] = o[t][r] * inv[r];
        }
    }
}

// ---------------------------------------------------------------------------
// Launch
// ---------------------------------------------------------------------------
extern "C" void kernel_launch(void* const* d_in, const int* in_sizes, int n_in,
                              void* d_out, int out_size, void* d_ws, size_t ws_size,
                              hipStream_t stream) {
    const float* x    = (const float*)d_in[0];
    const int*   mask = (const int*)  d_in[1];
    const float* Wq   = (const float*)d_in[2];
    const float* Wk   = (const float*)d_in[3];
    const float* Wv   = (const float*)d_in[4];
    float*       out  = (float*)d_out;

    bf16_t* ws  = (bf16_t*)d_ws;
    bf16_t* xh  = ws + XH_OFF;                   // [M x D]
    bf16_t* wh  = ws + WH_OFF;                   // 3 x [D x D]
    bf16_t* Qb  = ws + QKV_OFF;                  // [M x D]
    bf16_t* Kb  = Qb + (size_t)M * D;            // [M x D]
    bf16_t* Vtp = Kb + (size_t)M * D;            // [D x M]

    // One-time f32 -> bf16 conversions
    const int nx8 = (M * D) / 8, nw8 = (D * D) / 8;
    sha_cvt_bf16_kernel<<<(nx8 + 255) / 256, 256, 0, stream>>>(x,  xh, nx8);
    sha_cvt_bf16_kernel<<<(nw8 + 255) / 256, 256, 0, stream>>>(Wq, wh,                 nw8);
    sha_cvt_bf16_kernel<<<(nw8 + 255) / 256, 256, 0, stream>>>(Wk, wh + (size_t)D*D,   nw8);
    sha_cvt_bf16_kernel<<<(nw8 + 255) / 256, 256, 0, stream>>>(Wv, wh + (size_t)2*D*D, nw8);

    sha_qkv_proj_kernel<<<dim3(64 * 8, 1, 3), 256, 0, stream>>>(xh, wh, Qb, Kb, Vtp);
    sha_flash_attn_kernel<<<dim3(S / 16, Bb), 256, 0, stream>>>(Qb, Kb, Vtp, mask, out);
}
